// MyPVT2_52407190946105
// MI455X (gfx1250) — compile-verified, matrix operand order checked by reading, not run on hardware
//
#include <hip/hip_runtime.h>
#include <math.h>

// ---------------- problem constants ----------------
#define B_    8
#define NG_   256
#define N_    3136
#define C_    256
#define C2_   512
#define H_    8
#define S_    784
#define MLPH_ 2048
#define EPS_  1e-5f
#define DHEAD 64
#define NQK_  (NG_ + N_)   // 3392

typedef __bf16 bf16;
typedef bf16  v16bf __attribute__((ext_vector_type(16)));
typedef float v8f   __attribute__((ext_vector_type(8)));

union FragU { v16bf v; uint4 q[2]; };

// load a 16-bf16 fragment from two 16-byte-aligned chunks (matches CDNA5
// 16-bit A/B 16x32 layout: lane<16 -> K {0..7} u {16..23}, lane>=16 shifted)
__device__ __forceinline__ v16bf ld_frag(const bf16* p0, const bf16* p1) {
  FragU f;
  f.q[0] = *(const uint4*)p0;
  f.q[1] = *(const uint4*)p1;
  return f.v;
}

__device__ __forceinline__ v8f wmma_bf16f32(v16bf a, v16bf b, v8f c) {
  return __builtin_amdgcn_wmma_f32_16x16x32_bf16(
      /*neg_a=*/false, a, /*neg_b=*/false, b,
      /*c_mod=*/(short)0, c, /*reuse_a=*/false, /*reuse_b=*/false);
}

__device__ __forceinline__ float gelu_exact(float x) {
  return 0.5f * x * (1.0f + erff(x * 0.70710678118654752f));
}

// ---------------- confidence logits: LN(x_ada) @ conf_w + conf_b ----------------
__global__ void conf_logit_kernel(const float* __restrict__ xada,
                                  const float* __restrict__ g,
                                  const float* __restrict__ be,
                                  const float* __restrict__ cw,
                                  const float* __restrict__ cb,
                                  float* __restrict__ logits) {
  int row = blockIdx.x;                 // 0 .. B*N-1
  int tid = threadIdx.x;                // 256 == C_
  const float* x = xada + (long)row * C_;
  __shared__ float red[256];
  float v = x[tid];
  red[tid] = v; __syncthreads();
  for (int s = 128; s > 0; s >>= 1) { if (tid < s) red[tid] += red[tid + s]; __syncthreads(); }
  float mean = red[0] * (1.0f / C_);
  __syncthreads();
  float d = v - mean;
  red[tid] = d * d; __syncthreads();
  for (int s = 128; s > 0; s >>= 1) { if (tid < s) red[tid] += red[tid + s]; __syncthreads(); }
  float var = red[0] * (1.0f / C_);
  __syncthreads();
  float ln = d * rsqrtf(var + EPS_) * g[tid] + be[tid];
  red[tid] = ln * cw[tid]; __syncthreads();
  for (int s = 128; s > 0; s >>= 1) { if (tid < s) red[tid] += red[tid + s]; __syncthreads(); }
  if (tid == 0) logits[row] = red[0] + cb[0];
}

// ---------------- per-batch softmax over N, conf = softmax * N ----------------
__global__ void conf_softmax_kernel(const float* __restrict__ logits,
                                    float* __restrict__ conf) {
  int b = blockIdx.x, tid = threadIdx.x;
  const float* l = logits + (long)b * N_;
  __shared__ float red[256];
  float mx = -INFINITY;
  for (int i = tid; i < N_; i += 256) mx = fmaxf(mx, l[i]);
  red[tid] = mx; __syncthreads();
  for (int s = 128; s > 0; s >>= 1) { if (tid < s) red[tid] = fmaxf(red[tid], red[tid + s]); __syncthreads(); }
  mx = red[0];
  __syncthreads();
  float sm = 0.f;
  for (int i = tid; i < N_; i += 256) sm += expf(l[i] - mx);
  red[tid] = sm; __syncthreads();
  for (int s = 128; s > 0; s >>= 1) { if (tid < s) red[tid] += red[tid + s]; __syncthreads(); }
  float tot = red[0];
  for (int i = tid; i < N_; i += 256)
    conf[(long)b * N_ + i] = expf(l[i] - mx) / tot * (float)N_;
}

// ---------------- top-k (descending) via in-LDS bitonic sort of 4096 ----------------
__global__ void topk_kernel(const float* __restrict__ conf,
                            const int* __restrict__ pos_ada,
                            int* __restrict__ idx,
                            int* __restrict__ pos_down) {
  int b = blockIdx.x, tid = threadIdx.x;    // blockDim = 1024
  __shared__ float sv[4096];
  __shared__ int   si[4096];
  for (int i = tid; i < 4096; i += 1024) {
    sv[i] = (i < N_) ? conf[(long)b * N_ + i] : -INFINITY;
    si[i] = i;
  }
  __syncthreads();
  for (unsigned k = 2; k <= 4096; k <<= 1) {
    for (unsigned j = k >> 1; j >= 1; j >>= 1) {
      for (unsigned i = tid; i < 4096; i += 1024) {
        unsigned p = i ^ j;
        if (p > i) {
          bool descSeg = ((i & k) == 0);
          float a = sv[i], c = sv[p];
          int   ia = si[i], ic = si[p];
          bool swap = descSeg ? (a < c || (a == c && ia > ic))
                              : (a > c || (a == c && ia < ic));
          if (swap) { sv[i] = c; sv[p] = a; si[i] = ic; si[p] = ia; }
        }
      }
      __syncthreads();
    }
  }
  for (int s = tid; s < S_; s += 1024) {
    int id = si[s];
    idx[(long)b * S_ + s]      = id;
    pos_down[(long)b * S_ + s] = pos_ada[(long)b * N_ + id];
  }
}

// ---------------- gather x_down rows (f32 + bf16) ----------------
__global__ void gather_kernel(const float* __restrict__ xada,
                              const int* __restrict__ idx,
                              float* __restrict__ xdown,
                              bf16* __restrict__ xdownb) {
  long t = (long)blockIdx.x * blockDim.x + threadIdx.x;
  long total = (long)B_ * S_ * C_;
  if (t >= total) return;
  long bs = t / C_; int c = (int)(t - bs * C_);
  int b = (int)(bs / S_);
  int id = idx[bs];
  float v = xada[((long)b * N_ + id) * C_ + c];
  xdown[t]  = v;
  xdownb[t] = (bf16)v;
}

// ---------------- f32 -> bf16 cast ----------------
__global__ void cast_bf16_kernel(const float* __restrict__ in,
                                 bf16* __restrict__ out, long n) {
  long t = (long)blockIdx.x * blockDim.x + threadIdx.x;
  if (t < n) out[t] = (bf16)in[t];
}

// ---------------- f32 [K,Nc] -> bf16 transposed [Nc,K] ----------------
__global__ void transpose_cast_kernel(const float* __restrict__ w,
                                      bf16* __restrict__ wt, int K, int Nc) {
  long t = (long)blockIdx.x * blockDim.x + threadIdx.x;
  long total = (long)K * Nc;
  if (t >= total) return;
  long n = t / K; int k = (int)(t - n * K);
  wt[t] = (bf16)w[(long)k * Nc + n];
}

// ---------------- LayerNorm (optional per-row pre-scale) -> bf16 ----------------
__global__ void ln_bf16_kernel(const float* __restrict__ in, long inb, int rpb,
                               bf16* __restrict__ out, long outb, int C,
                               const float* __restrict__ g,
                               const float* __restrict__ be,
                               const float* __restrict__ scale) {
  int row = blockIdx.x;
  int b = row / rpb, r = row - b * rpb;
  const float* x = in + (long)b * inb + (long)r * C;
  bf16* o = out + (long)b * outb + (long)r * C;
  float sc = scale ? scale[(long)b * rpb + r] : 1.0f;
  int tid = threadIdx.x;                 // 256
  __shared__ float red[256];
  float lsum = 0.f, lsq = 0.f;
  for (int c = tid; c < C; c += 256) { float v = x[c] * sc; lsum += v; lsq += v * v; }
  red[tid] = lsum; __syncthreads();
  for (int s = 128; s > 0; s >>= 1) { if (tid < s) red[tid] += red[tid + s]; __syncthreads(); }
  float mean = red[0] / (float)C;
  __syncthreads();
  red[tid] = lsq; __syncthreads();
  for (int s = 128; s > 0; s >>= 1) { if (tid < s) red[tid] += red[tid + s]; __syncthreads(); }
  float var = red[0] / (float)C - mean * mean;
  float inv = rsqrtf(var + EPS_);
  for (int c = tid; c < C; c += 256) {
    float v = x[c] * sc;
    o[c] = (bf16)((v - mean) * inv * g[c] + be[c]);
  }
}

// ---------------- WMMA GEMM: out = A(bf16)[M,K] @ W (WT bf16 [Nc,K]) ----------
// Block: 256 threads = 8 waves; tile 128x64; BK=64; wave tile 32x32 (2x2 frags).
// Register double-buffered global->LDS staging; padded LDS rows (no bank dup).
#define APAD 72   // 128x64 tile, row stride 72 bf16 (144B) -> distinct banks
#define BPAD 72
__global__ __launch_bounds__(256)
void wmma_gemm_kernel(const bf16* __restrict__ A, long a_bstr, int rpb, int K,
                      const bf16* __restrict__ WT, int Nc,
                      const float* __restrict__ bias, int do_gelu,
                      const float* __restrict__ res,
                      float* __restrict__ outF, bf16* __restrict__ outB) {
  __shared__ __align__(16) bf16 As[128][APAD];
  __shared__ __align__(16) bf16 Bs[64][BPAD];

  int tid = threadIdx.x;
  int m0 = blockIdx.y * 128;
  int n0 = blockIdx.x * 64;
  int lane = tid & 31, wid = tid >> 5;
  int wm = wid >> 1, wn = wid & 1;
  int half = lane >> 4, l15 = lane & 15;

  v8f zero8 = {0.f, 0.f, 0.f, 0.f, 0.f, 0.f, 0.f, 0.f};
  v8f acc[2][2] = {{zero8, zero8}, {zero8, zero8}};

  // A loader: 2 chunks of 8 bf16 (16 B) per thread, 16-col steps (128 rows x 64 cols)
  int c0 = tid * 2, c1 = tid * 2 + 1;
  int ar0 = c0 >> 2, ac0 = (c0 & 3) * 16;
  int ar1 = c1 >> 2, ac1 = (c1 & 3) * 16;
  int g0 = m0 + ar0; int ab0 = g0 / rpb;
  int g1 = m0 + ar1; int ab1 = g1 / rpb;
  const bf16* ap0 = A + (long)ab0 * a_bstr + (long)(g0 - ab0 * rpb) * K + ac0;
  const bf16* ap1 = A + (long)ab1 * a_bstr + (long)(g1 - ab1 * rpb) * K + ac1;

  // B loader: 1 chunk of 8 bf16 (16 B) per thread, 16-col steps (64 n-rows x 64 k-cols of WT)
  int bn = tid >> 2, bc = (tid & 3) * 16;
  const bf16* bp = WT + (long)(n0 + bn) * K + bc;

  uint4 aR0 = *(const uint4*)ap0;
  uint4 aR1 = *(const uint4*)ap1;
  uint4 bR  = *(const uint4*)bp;

  for (int k0 = 0;;) {
    *(uint4*)&As[ar0][ac0] = aR0;
    *(uint4*)&As[ar1][ac1] = aR1;
    *(uint4*)&Bs[bn][bc]   = bR;
    __syncthreads();

    int k1 = k0 + 64;
    bool more = (k1 < K);
    uint4 nA0 = aR0, nA1 = aR1, nB = bR;
    if (more) {                                   // in flight under the WMMAs
      nA0 = *(const uint4*)(ap0 + k1);
      nA1 = *(const uint4*)(ap1 + k1);
      nB  = *(const uint4*)(bp + k1);
      if (k1 + 64 < K) {
        __builtin_prefetch(ap0 + k1 + 64, 0, 3);  // global_prefetch_b8
        __builtin_prefetch(bp + k1 + 64, 0, 3);
      }
    }

#pragma unroll
    for (int kk = 0; kk < 64; kk += 32) {
      v16bf aF0 = ld_frag(&As[wm * 32 + l15][kk + half * 8],
                          &As[wm * 32 + l15][kk + half * 8 + 16]);
      v16bf aF1 = ld_frag(&As[wm * 32 + 16 + l15][kk + half * 8],
                          &As[wm * 32 + 16 + l15][kk + half * 8 + 16]);
      v16bf bF0 = ld_frag(&Bs[wn * 32 + l15][kk + half * 8],
                          &Bs[wn * 32 + l15][kk + half * 8 + 16]);
      v16bf bF1 = ld_frag(&Bs[wn * 32 + 16 + l15][kk + half * 8],
                          &Bs[wn * 32 + 16 + l15][kk + half * 8 + 16]);
      acc[0][0] = wmma_bf16f32(aF0, bF0, acc[0][0]);
      acc[0][1] = wmma_bf16f32(aF0, bF1, acc[0][1]);
      acc[1][0] = wmma_bf16f32(aF1, bF0, acc[1][0]);
      acc[1][1] = wmma_bf16f32(aF1, bF1, acc[1][1]);
    }
    __syncthreads();
    if (!more) break;
    aR0 = nA0; aR1 = nA1; bR = nB; k0 = k1;
  }

  for (int mi = 0; mi < 2; mi++) {
    for (int ni = 0; ni < 2; ni++) {
      int col = n0 + wn * 32 + ni * 16 + l15;
      float bv = bias ? bias[col] : 0.f;
      for (int r = 0; r < 8; r++) {
        int row = m0 + wm * 32 + mi * 16 + half * 8 + r;
        float val = acc[mi][ni][r] + bv;
        if (do_gelu) val = gelu_exact(val);
        long off = (long)row * Nc + col;
        if (res)  val += res[off];
        if (outF) outF[off] = val;
        if (outB) outB[off] = (bf16)val;
      }
    }
  }
}

// ---------------- single-wave flash attention (d=64, 32 q-rows / block) --------
__global__ __launch_bounds__(32)
void flash_attn_kernel(const bf16* __restrict__ Q, const bf16* __restrict__ Kb,
                       const bf16* __restrict__ Vb, bf16* __restrict__ O,
                       int Nq, int Nk, float scale) {
  int lane = threadIdx.x;
  int nqT = (Nq + 31) / 32;
  int qt = blockIdx.x % nqT;
  int h  = (blockIdx.x / nqT) % H_;
  int b  = blockIdx.x / (nqT * H_);
  int q0 = qt * 32;
  int half = lane >> 4, l15 = lane & 15;

  __shared__ __align__(16) bf16 qt_s[32][72];   // 64 used, padded rows
  __shared__ __align__(16) bf16 p_s[32][40];    // 32 used
  __shared__ __align__(16) bf16 vt_s[64][40];   // 32 used

  { // stage q tile (one row per lane, 128B)
    int row = q0 + lane;
    uint4* dst = (uint4*)&qt_s[lane][0];
    if (row < Nq) {
      const uint4* src = (const uint4*)(Q + ((long)(b * Nq + row)) * C2_ + h * DHEAD);
      for (int i = 0; i < 8; i++) dst[i] = src[i];
    } else {
      uint4 z = {0u, 0u, 0u, 0u};
      for (int i = 0; i < 8; i++) dst[i] = z;
    }
  }
  __syncthreads();

  v8f zero8 = {0.f, 0.f, 0.f, 0.f, 0.f, 0.f, 0.f, 0.f};
  v8f o[2][4];
  float ms[2][8], ls[2][8];
  for (int mi = 0; mi < 2; mi++) {
    for (int n2 = 0; n2 < 4; n2++) o[mi][n2] = zero8;
    for (int r = 0; r < 8; r++) { ms[mi][r] = -INFINITY; ls[mi][r] = 0.f; }
  }

  int nkT = Nk / 32;
  for (int kt = 0; kt < nkT; kt++) {
    { // stage V^T tile: vt_s[d][key]
      int key = kt * 32 + lane;
      const bf16* vsrc = Vb + ((long)(b * Nk + key)) * C2_ + h * DHEAD;
      for (int d = 0; d < DHEAD; d++) vt_s[d][lane] = vsrc[d];
    }

    // S = Q @ K^T  (K fragments straight from global: rows are K-contiguous)
    v8f sf[2][2] = {{zero8, zero8}, {zero8, zero8}};
    for (int kc = 0; kc < 2; kc++) {
      v16bf a0 = ld_frag(&qt_s[l15][kc * 32 + half * 8],
                         &qt_s[l15][kc * 32 + half * 8 + 16]);
      v16bf a1 = ld_frag(&qt_s[16 + l15][kc * 32 + half * 8],
                         &qt_s[16 + l15][kc * 32 + half * 8 + 16]);
      for (int ni = 0; ni < 2; ni++) {
        int key = kt * 32 + ni * 16 + l15;
        const bf16* kp = Kb + ((long)(b * Nk + key)) * C2_ + h * DHEAD + kc * 32 + half * 8;
        v16bf bF = ld_frag(kp, kp + 16);
        sf[0][ni] = wmma_bf16f32(a0, bF, sf[0][ni]);
        sf[1][ni] = wmma_bf16f32(a1, bF, sf[1][ni]);
      }
    }
    __syncthreads();

    // online softmax (rows striped over C-fragment VGPR slots, per 16-lane half)
    for (int mi = 0; mi < 2; mi++) {
      for (int r = 0; r < 8; r++) {
        float s0 = sf[mi][0][r] * scale;
        float s1 = sf[mi][1][r] * scale;
        float tmax = fmaxf(s0, s1);
        for (int off = 1; off <= 8; off <<= 1)
          tmax = fmaxf(tmax, __shfl_xor(tmax, off, 32));
        float mn = fmaxf(ms[mi][r], tmax);
        float corr = expf(ms[mi][r] - mn);
        float p0 = expf(s0 - mn);
        float p1 = expf(s1 - mn);
        p_s[mi * 16 + half * 8 + r][l15]      = (bf16)p0;
        p_s[mi * 16 + half * 8 + r][16 + l15] = (bf16)p1;
        float ts = p0 + p1;
        for (int off = 1; off <= 8; off <<= 1)
          ts += __shfl_xor(ts, off, 32);
        ls[mi][r] = ls[mi][r] * corr + ts;
        ms[mi][r] = mn;
        for (int n2 = 0; n2 < 4; n2++) o[mi][n2][r] *= corr;
      }
    }
    __syncthreads();

    // O += P @ V
    {
      v16bf bv0 = ld_frag(&vt_s[l15][half * 8],      &vt_s[l15][half * 8 + 16]);
      v16bf bv1 = ld_frag(&vt_s[16 + l15][half * 8], &vt_s[16 + l15][half * 8 + 16]);
      v16bf bv2 = ld_frag(&vt_s[32 + l15][half * 8], &vt_s[32 + l15][half * 8 + 16]);
      v16bf bv3 = ld_frag(&vt_s[48 + l15][half * 8], &vt_s[48 + l15][half * 8 + 16]);
      for (int mi = 0; mi < 2; mi++) {
        v16bf ap = ld_frag(&p_s[mi * 16 + l15][half * 8],
                           &p_s[mi * 16 + l15][half * 8 + 16]);
        o[mi][0] = wmma_bf16f32(ap, bv0, o[mi][0]);
        o[mi][1] = wmma_bf16f32(ap, bv1, o[mi][1]);
        o[mi][2] = wmma_bf16f32(ap, bv2, o[mi][2]);
        o[mi][3] = wmma_bf16f32(ap, bv3, o[mi][3]);
      }
    }
    __syncthreads();
  }

  for (int mi = 0; mi < 2; mi++) {
    for (int r = 0; r < 8; r++) {
      int row = q0 + mi * 16 + half * 8 + r;
      if (row >= Nq) continue;
      float inv = 1.f / ls[mi][r];
      bf16* op = O + ((long)(b * Nq + row)) * C2_ + h * DHEAD + l15;
      for (int n2 = 0; n2 < 4; n2++)
        op[n2 * 16] = (bf16)(o[mi][n2][r] * inv);
    }
  }
}

// ---------------- pos-embed gather-add ----------------
__global__ void pos_add_kernel(float* __restrict__ out,
                               const float* __restrict__ pe,
                               const int* __restrict__ pos, long rows) {
  long t = (long)blockIdx.x * blockDim.x + threadIdx.x;
  long total = rows * C2_;
  if (t >= total) return;
  long rr = t / C2_; int c = (int)(t - rr * C2_);
  out[t] += pe[(long)pos[rr] * C2_ + c];
}

// =======================================================================
extern "C" void kernel_launch(void* const* d_in, const int* in_sizes, int n_in,
                              void* d_out, int out_size, void* d_ws, size_t ws_size,
                              hipStream_t stream) {
  const float* x_grid    = (const float*)d_in[0];
  const float* x_ada     = (const float*)d_in[1];
  const int*   pos_grid  = (const int*)d_in[2];
  const int*   pos_ada   = (const int*)d_in[3];
  const float* pos_embed = (const float*)d_in[4];
  const float* norm_g = (const float*)d_in[5];
  const float* norm_b = (const float*)d_in[6];
  const float* conf_w = (const float*)d_in[7];
  const float* conf_b = (const float*)d_in[8];
  const float* n1_g = (const float*)d_in[9];
  const float* n1_b = (const float*)d_in[10];
  const float* q_w  = (const float*)d_in[11];
  const float* k_w  = (const float*)d_in[12];
  const float* v_w  = (const float*)d_in[13];
  const float* k2_w = (const float*)d_in[14];
  const float* v2_w = (const float*)d_in[15];
  const float* proj_w = (const float*)d_in[16];
  const float* proj_b = (const float*)d_in[17];
  const float* fc_w = (const float*)d_in[18];
  const float* fc_b = (const float*)d_in[19];
  const float* n2_g = (const float*)d_in[20];
  const float* n2_b = (const float*)d_in[21];
  const float* mlp1_w = (const float*)d_in[22];
  const float* mlp1_b = (const float*)d_in[23];
  const float* mlp2_w = (const float*)d_in[24];
  const float* mlp2_b = (const float*)d_in[25];

  char* ws = (char*)d_ws;
  size_t off = 0;
  auto alloc = [&](size_t bytes) -> char* {
    char* p = ws + off;
    off += (bytes + 255) & ~(size_t)255;
    return p;
  };

  float* conf_logits = (float*)alloc((size_t)B_ * N_ * 4);
  float* conf        = (float*)alloc((size_t)B_ * N_ * 4);
  int*   topk_idx    = (int*)  alloc((size_t)B_ * S_ * 4);
  int*   pos_down    = (int*)  alloc((size_t)B_ * S_ * 4);
  float* x_down      = (float*)alloc((size_t)B_ * S_ * C_ * 4);
  bf16*  xdown_b     = (bf16*) alloc((size_t)B_ * S_ * C_ * 2);
  bf16*  xgrid_b     = (bf16*) alloc((size_t)B_ * NG_ * C_ * 2);
  bf16*  src         = (bf16*) alloc((size_t)B_ * NQK_ * C_ * 2);
  bf16*  lnxd        = (bf16*) alloc((size_t)B_ * S_ * C_ * 2);
  bf16*  kbuf        = (bf16*) alloc((size_t)B_ * NQK_ * C2_ * 2);
  bf16*  vbuf        = (bf16*) alloc((size_t)B_ * NQK_ * C2_ * 2);
  bf16*  qg          = (bf16*) alloc((size_t)B_ * NG_ * C2_ * 2);
  bf16*  qa          = (bf16*) alloc((size_t)B_ * S_ * C2_ * 2);
  bf16*  att1        = (bf16*) alloc((size_t)B_ * NG_ * C2_ * 2);
  float* yg          = (float*)alloc((size_t)B_ * NG_ * C2_ * 4);
  bf16*  yg_b        = (bf16*) alloc((size_t)B_ * NG_ * C2_ * 2);
  bf16*  k2buf       = (bf16*) alloc((size_t)B_ * NG_ * C2_ * 2);
  bf16*  v2buf       = (bf16*) alloc((size_t)B_ * NG_ * C2_ * 2);
  bf16*  att2        = (bf16*) alloc((size_t)B_ * S_ * C2_ * 2);
  float* ya          = (float*)alloc((size_t)B_ * S_ * C2_ * 4);
  bf16*  lng         = (bf16*) alloc((size_t)B_ * NG_ * C2_ * 2);
  bf16*  lna         = (bf16*) alloc((size_t)B_ * S_ * C2_ * 2);
  bf16*  hg          = (bf16*) alloc((size_t)B_ * NG_ * MLPH_ * 2);
  bf16*  ha          = (bf16*) alloc((size_t)B_ * S_ * MLPH_ * 2);
  // transposed bf16 weights [Nc][K]
  bf16* q_wt    = (bf16*)alloc((size_t)C_ * C2_ * 2);
  bf16* k_wt    = (bf16*)alloc((size_t)C_ * C2_ * 2);
  bf16* v_wt    = (bf16*)alloc((size_t)C_ * C2_ * 2);
  bf16* k2_wt   = (bf16*)alloc((size_t)C2_ * C2_ * 2);
  bf16* v2_wt   = (bf16*)alloc((size_t)C2_ * C2_ * 2);
  bf16* proj_wt = (bf16*)alloc((size_t)C2_ * C2_ * 2);
  bf16* fc_wt   = (bf16*)alloc((size_t)C_ * C2_ * 2);
  bf16* mlp1_wt = (bf16*)alloc((size_t)C2_ * MLPH_ * 2);
  bf16* mlp2_wt = (bf16*)alloc((size_t)MLPH_ * C2_ * 2);
  (void)ws_size; (void)in_sizes; (void)n_in; (void)out_size;

  const float scale = 0.17677669529663688f;  // (C/H)^-0.5 = 1/sqrt(32)

  auto xpose = [&](const float* w, bf16* wt, int K, int Nc) {
    long tot = (long)K * Nc;
    transpose_cast_kernel<<<(tot + 255) / 256, 256, 0, stream>>>(w, wt, K, Nc);
  };
  xpose(q_w,    q_wt,    C_,    C2_);
  xpose(k_w,    k_wt,    C_,    C2_);
  xpose(v_w,    v_wt,    C_,    C2_);
  xpose(k2_w,   k2_wt,   C2_,   C2_);
  xpose(v2_w,   v2_wt,   C2_,   C2_);
  xpose(proj_w, proj_wt, C2_,   C2_);
  xpose(fc_w,   fc_wt,   C_,    C2_);
  xpose(mlp1_w, mlp1_wt, C2_,   MLPH_);
  xpose(mlp2_w, mlp2_wt, MLPH_, C2_);

  // ---- DownLayer2 ----
  conf_logit_kernel<<<B_ * N_, 256, 0, stream>>>(x_ada, norm_g, norm_b, conf_w, conf_b, conf_logits);
  conf_softmax_kernel<<<B_, 256, 0, stream>>>(conf_logits, conf);
  topk_kernel<<<B_, 1024, 0, stream>>>(conf, pos_ada, topk_idx, pos_down);
  {
    long tot = (long)B_ * S_ * C_;
    gather_kernel<<<(tot + 255) / 256, 256, 0, stream>>>(x_ada, topk_idx, x_down, xdown_b);
  }
  {
    long tot = (long)B_ * NG_ * C_;
    cast_bf16_kernel<<<(tot + 255) / 256, 256, 0, stream>>>(x_grid, xgrid_b, tot);
  }

  // ---- layer norms feeding attention ----
  ln_bf16_kernel<<<B_ * NG_, 256, 0, stream>>>(x_grid, (long)NG_ * C_, NG_,
                                               src, (long)NQK_ * C_, C_, n1_g, n1_b, nullptr);
  ln_bf16_kernel<<<B_ * N_, 256, 0, stream>>>(x_ada, (long)N_ * C_, N_,
                                              src + (long)NG_ * C_, (long)NQK_ * C_, C_, n1_g, n1_b, conf);
  ln_bf16_kernel<<<B_ * S_, 256, 0, stream>>>(x_down, (long)S_ * C_, S_,
                                              lnxd, (long)S_ * C_, C_, n1_g, n1_b, nullptr);

  auto gemm = [&](const bf16* A, long abstr, int rpb, int K,
                  const bf16* WT, int Nc, int M,
                  const float* bias, int gelu, const float* res,
                  float* oF, bf16* oB) {
    dim3 g(Nc / 64, M / 128);
    wmma_gemm_kernel<<<g, 256, 0, stream>>>(A, abstr, rpb, K, WT, Nc, bias, gelu, res, oF, oB);
  };

  // ---- attention 1: grid attends to cat(grid, ada) ----
  gemm(src, (long)NQK_ * C_, NQK_, C_, k_wt, C2_, B_ * NQK_, nullptr, 0, nullptr, nullptr, kbuf);
  gemm(src, (long)NQK_ * C_, NQK_, C_, v_wt, C2_, B_ * NQK_, nullptr, 0, nullptr, nullptr, vbuf);
  gemm(src, (long)NQK_ * C_, NG_,  C_, q_wt, C2_, B_ * NG_,  nullptr, 0, nullptr, nullptr, qg);
  flash_attn_kernel<<<B_ * H_ * ((NG_ + 31) / 32), 32, 0, stream>>>(qg, kbuf, vbuf, att1, NG_, NQK_, scale);
  gemm(att1, (long)NG_ * C2_, NG_, C2_, proj_wt, C2_, B_ * NG_, proj_b, 0, nullptr, yg, yg_b);

  // ---- attention 2: downsampled tokens attend to yg ----
  gemm(lnxd, (long)S_ * C_, S_, C_, q_wt, C2_, B_ * S_, nullptr, 0, nullptr, nullptr, qa);
  gemm(yg_b, (long)NG_ * C2_, NG_, C2_, k2_wt, C2_, B_ * NG_, nullptr, 0, nullptr, nullptr, k2buf);
  gemm(yg_b, (long)NG_ * C2_, NG_, C2_, v2_wt, C2_, B_ * NG_, nullptr, 0, nullptr, nullptr, v2buf);
  flash_attn_kernel<<<B_ * H_ * ((S_ + 31) / 32), 32, 0, stream>>>(qa, k2buf, v2buf, att2, S_, NG_, scale);
  gemm(att2, (long)S_ * C2_, S_, C2_, proj_wt, C2_, B_ * S_, proj_b, 0, nullptr, ya, nullptr);

  // ---- fc residual into d_out ----
  float* out_xg = (float*)d_out;
  float* out_xa = out_xg + (size_t)B_ * NG_ * C2_;
  gemm(xgrid_b, (long)NG_ * C_, NG_, C_, fc_wt, C2_, B_ * NG_, fc_b, 0, yg, out_xg, nullptr);
  gemm(xdown_b, (long)S_ * C_, S_, C_, fc_wt, C2_, B_ * S_, fc_b, 0, ya, out_xa, nullptr);

  // ---- MLP (LN -> 2048 GELU -> 512) + residual, in place on d_out ----
  ln_bf16_kernel<<<B_ * NG_, 256, 0, stream>>>(out_xg, (long)NG_ * C2_, NG_,
                                               lng, (long)NG_ * C2_, C2_, n2_g, n2_b, nullptr);
  ln_bf16_kernel<<<B_ * S_, 256, 0, stream>>>(out_xa, (long)S_ * C2_, S_,
                                              lna, (long)S_ * C2_, C2_, n2_g, n2_b, nullptr);
  gemm(lng, (long)NG_ * C2_, NG_, C2_, mlp1_wt, MLPH_, B_ * NG_, mlp1_b, 1, nullptr, nullptr, hg);
  gemm(lna, (long)S_ * C2_, S_, C2_, mlp1_wt, MLPH_, B_ * S_, mlp1_b, 1, nullptr, nullptr, ha);
  gemm(hg, (long)NG_ * MLPH_, NG_, MLPH_, mlp2_wt, C2_, B_ * NG_, mlp2_b, 0, out_xg, out_xg, nullptr);
  gemm(ha, (long)S_ * MLPH_, S_, MLPH_, mlp2_wt, C2_, B_ * S_, mlp2_b, 0, out_xa, out_xa, nullptr);

  // ---- positional embedding adds ----
  {
    long tot = (long)B_ * NG_ * C2_;
    pos_add_kernel<<<(tot + 255) / 256, 256, 0, stream>>>(out_xg, pos_embed, pos_grid, (long)B_ * NG_);
  }
  {
    long tot = (long)B_ * S_ * C2_;
    pos_add_kernel<<<(tot + 255) / 256, 256, 0, stream>>>(out_xa, pos_embed, pos_down, (long)B_ * S_);
  }
}
